// SpatialTransformer_47364899340562
// MI455X (gfx1250) — matrix operand order repaired and checked by reading, not compiled, not run on hardware
//
#include <hip/hip_runtime.h>
#include <math.h>

typedef __bf16 bf16;
typedef __attribute__((ext_vector_type(16))) __bf16 v16bf;
typedef __attribute__((ext_vector_type(8)))  __bf16 v8bf;
typedef __attribute__((ext_vector_type(8)))  float  v8f;

#define TOKENS 4096   // B*H*W
#define CH     512
#define NHEAD  16
#define DHEAD  32
#define SEQ    1024   // H*W
#define NLAYER 4
#define DFFH   2048   // DFF

#define BS_PITCH 48   // padded LDS row pitch (bf16 elems) to reduce bank conflicts

union BF16Frag { v16bf v; v8bf h[2]; };

// ---- WMMA fragment loaders (wave32; contiguous k = 16*hf + s, same order in A and B) ----
// A (16x32 bf16, row-major source): lane<16 holds M=lane, K=k0..k0+15;
// lane>=16 holds M=lane-16, K=k0+16..k0+31.
static __device__ inline v16bf load_a_frag(const bf16* A, int lda, int row, int k0, int hf) {
  BF16Frag u;
  const bf16* p = A + (size_t)row * lda + k0 + hf * 16;
  u.h[0] = *(const v8bf*)(p);
  u.h[1] = *(const v8bf*)(p + 8);
  return u.v;
}
// B (32x16 bf16) sourced from B^T [N,K] row-major: lane<16 -> N=lane, K=k0..k0+15;
// lane>=16 -> N=lane-16, K=k0+16..k0+31. 16 contiguous bf16 per lane.
static __device__ inline v16bf load_bt_frag(const bf16* Bt, int ldbt, int col, int k0, int hf) {
  BF16Frag u;
  const bf16* p = Bt + (size_t)col * ldbt + k0 + hf * 16;
  u.h[0] = *(const v8bf*)(p);
  u.h[1] = *(const v8bf*)(p + 8);
  return u.v;
}
// B fragment from LDS tile staged as [64 cols][BS_PITCH k] (32 valid k per row).
static __device__ inline v16bf load_b_frag_lds(const bf16* Bs, int col, int hf) {
  BF16Frag u;
  const bf16* p = Bs + col * BS_PITCH + hf * 16;
  u.h[0] = *(const v8bf*)(p);
  u.h[1] = *(const v8bf*)(p + 8);
  return u.v;
}
static __device__ inline v8f wmma_bf16(v16bf a, v16bf b, v8f c) {
  return __builtin_amdgcn_wmma_f32_16x16x32_bf16(false, a, false, b, (short)0, c, false, false);
}

// ---- CDNA5 async global->LDS copy (per-lane 16B), tracked by ASYNCcnt ----
static __device__ inline void async_copy_b128(unsigned lds_off, const bf16* gptr) {
  asm volatile("global_load_async_to_lds_b128 %0, %1, off"
               :: "v"(lds_off), "v"(gptr) : "memory");
}

// ---- weight transpose + f32->bf16:  in [batch][K][N] f32  ->  out [batch][N][K] bf16 ----
__global__ __launch_bounds__(256) void convert_wT_kernel(const float* __restrict__ in,
                                                         bf16* __restrict__ out,
                                                         int K, int N, long total) {
  long idx = (long)blockIdx.x * 256 + threadIdx.x;
  if (idx >= total) return;
  long per = (long)K * N;
  long bb = idx / per;
  long r  = idx % per;
  long n  = r / K;
  long kk = r % K;
  out[idx] = (bf16)in[bb * per + kk * N + n];
}

__global__ __launch_bounds__(256) void f32_to_bf16_kernel(const float* __restrict__ in,
                                                          bf16* __restrict__ out, int n) {
  int i = blockIdx.x * 256 + threadIdx.x;
  if (i < n) out[i] = (bf16)in[i];
}

// ---- GroupNorm: x [B,C,H,W] -> token-major bf16 [B*HW, C] with affine ----
__global__ __launch_bounds__(256) void groupnorm_kernel(const float* __restrict__ x,
                                                        const float* __restrict__ w,
                                                        const float* __restrict__ b,
                                                        bf16* __restrict__ out) {
  int bz = blockIdx.x >> 4, g = blockIdx.x & 15;     // 16 groups, 32 ch each
  const float* xp = x + ((size_t)bz * CH + g * 32) * SEQ;
  __shared__ float rs[256], rq[256];
  float s = 0.f, q = 0.f;
  for (int e = threadIdx.x; e < 32 * SEQ; e += 256) { float v = xp[e]; s += v; q += v * v; }
  rs[threadIdx.x] = s; rq[threadIdx.x] = q;
  __syncthreads();
  for (int off = 128; off > 0; off >>= 1) {
    if (threadIdx.x < off) { rs[threadIdx.x] += rs[threadIdx.x + off]; rq[threadIdx.x] += rq[threadIdx.x + off]; }
    __syncthreads();
  }
  float mean = rs[0] / (32.f * SEQ);
  float var  = rq[0] / (32.f * SEQ) - mean * mean;
  float rstd = rsqrtf(var + 1e-6f);
  for (int e = threadIdx.x; e < 32 * SEQ; e += 256) {
    int c = g * 32 + (e >> 10);
    int p = e & (SEQ - 1);
    float v = (xp[e] - mean) * rstd * w[c] + b[c];
    out[((size_t)bz * SEQ + p) * CH + c] = (bf16)v;
  }
}

// ---- LayerNorm over C=512: X f32 [T,512] -> bf16 [T,512] ----
__global__ __launch_bounds__(128) void layernorm_kernel(const float* __restrict__ X,
                                                        const float* __restrict__ w,
                                                        const float* __restrict__ b,
                                                        bf16* __restrict__ out) {
  int t = blockIdx.x;
  const float* xp = X + (size_t)t * CH;
  __shared__ float rs[128], rq[128];
  float s = 0.f, q = 0.f;
  for (int c = threadIdx.x; c < CH; c += 128) { float v = xp[c]; s += v; q += v * v; }
  rs[threadIdx.x] = s; rq[threadIdx.x] = q;
  __syncthreads();
  for (int off = 64; off > 0; off >>= 1) {
    if (threadIdx.x < off) { rs[threadIdx.x] += rs[threadIdx.x + off]; rq[threadIdx.x] += rq[threadIdx.x + off]; }
    __syncthreads();
  }
  float mean = rs[0] / (float)CH;
  float var  = rq[0] / (float)CH - mean * mean;
  float rstd = rsqrtf(var + 1e-5f);
  for (int c = threadIdx.x; c < CH; c += 128)
    out[(size_t)t * CH + c] = (bf16)((xp[c] - mean) * rstd * w[c] + b[c]);
}

// ---- GEMM epilogue (one 16x16 accumulator tile) ----
// MODE 0: outf[m*N+n] = acc + bias[n]                       (conv-in -> X)
// MODE 1: q/k per-head scatter bf16 [B,NH,SEQ,DH]
// MODE 2: v transposed scatter bf16 [B,NH,DH,SEQ]
// MODE 3: outf[m*N+n] += acc + bias[n]                      (residual accumulate into X)
// MODE 4: conv-out: out[b,c,p] = acc + bias + resid[b,c,p]  (NCHW scatter)
template <int MODE>
static __device__ inline void epilogue_tile(const v8f& acc, int mbase, int col,
                                            const float* bias, float* outf, bf16* outb,
                                            const float* resid, int N, int hf) {
#pragma unroll
  for (int r = 0; r < 8; ++r) {
    int m = mbase + r + 8 * hf;               // C/D layout: VGPR r -> M=r (lanes<16), r+8 (lanes>=16)
    float v = acc[r];
    if constexpr (MODE == 0) {
      outf[(size_t)m * N + col] = v + bias[col];
    } else if constexpr (MODE == 1) {
      int b = m >> 10, i = m & (SEQ - 1);
      int h = col >> 5, d = col & 31;
      outb[(((size_t)b * NHEAD + h) * SEQ + i) * DHEAD + d] = (bf16)v;
    } else if constexpr (MODE == 2) {
      int b = m >> 10, i = m & (SEQ - 1);
      int h = col >> 5, d = col & 31;
      outb[(((size_t)b * NHEAD + h) * DHEAD + d) * SEQ + i] = (bf16)v;
    } else if constexpr (MODE == 3) {
      outf[(size_t)m * N + col] += v + bias[col];
    } else if constexpr (MODE == 4) {
      int b = m >> 10, p = m & (SEQ - 1);
      size_t i0 = ((size_t)b * CH + col) * SEQ + p;
      outf[i0] = v + bias[col] + resid[i0];
    }
  }
}

// ---- bf16 WMMA GEMM: D = A[M,K] * B[K,N] (+epilogue). B given transposed [N,K]. ----
// 256 threads = 8 waves (4 in M x 2 in N); wave tile 32x32 (4 WMMAs/K-step);
// block tile 128x64. B tile (64x32 = 4KB/step) staged to LDS via
// global_load_async_to_lds_b128 (1 per thread per step), double-buffered,
// synchronized with s_wait_asynccnt + barrier. A loaded directly (2x reuse in block).
template <int MODE>
__global__ __launch_bounds__(256) void gemm_bf16_kernel(const bf16* __restrict__ A,
                                                        const bf16* __restrict__ Bt,
                                                        const float* __restrict__ bias,
                                                        float* __restrict__ outf,
                                                        bf16* __restrict__ outb,
                                                        const float* __restrict__ resid,
                                                        int N, int K, int lda, int ldbt) {
  __shared__ __align__(16) bf16 Bs[2][64 * BS_PITCH];
  int tid = threadIdx.x;
  int lane = tid & 31, wave = tid >> 5;
  int hf = lane >> 4, l15 = lane & 15;
  int wm = wave >> 1, wn = wave & 1;
  int m0 = blockIdx.x * 128 + wm * 32;
  int nblk = blockIdx.y * 64;
  int n0 = nblk + wn * 32;

  // staging map: thread -> (col 0..63, 16B segment 0..3) of the 64x32 B tile
  int scol = tid >> 2, sseg = tid & 3;
  const bf16* sg_base = Bt + (size_t)(nblk + scol) * ldbt + sseg * 8;
  unsigned slds0 = (unsigned)(uintptr_t)(&Bs[0][scol * BS_PITCH + sseg * 8]);
  unsigned slds1 = (unsigned)(uintptr_t)(&Bs[1][scol * BS_PITCH + sseg * 8]);

  v8f acc00 = {}; v8f acc01 = {}; v8f acc10 = {}; v8f acc11 = {};
  async_copy_b128(slds0, sg_base);                       // stage k0 = 0 into buf 0
  int buf = 0;
  for (int k0 = 0; k0 < K; k0 += 32) {
    if (k0 + 32 < K) {
      async_copy_b128(buf ? slds0 : slds1, sg_base + k0 + 32);  // prefetch next
      asm volatile("s_wait_asynccnt 0x1" ::: "memory");         // current buffer done
    } else {
      asm volatile("s_wait_asynccnt 0x0" ::: "memory");
    }
    __syncthreads();
    v16bf a0 = load_a_frag(A, lda, m0 + l15,      k0, hf);
    v16bf a1 = load_a_frag(A, lda, m0 + 16 + l15, k0, hf);
    const bf16* bsrc = &Bs[buf][0];
    v16bf b0 = load_b_frag_lds(bsrc, wn * 32 + l15,      hf);
    v16bf b1 = load_b_frag_lds(bsrc, wn * 32 + 16 + l15, hf);
    acc00 = wmma_bf16(a0, b0, acc00);
    acc01 = wmma_bf16(a0, b1, acc01);
    acc10 = wmma_bf16(a1, b0, acc10);
    acc11 = wmma_bf16(a1, b1, acc11);
    __syncthreads();                                     // all waves done reading buf
    buf ^= 1;
  }
  epilogue_tile<MODE>(acc00, m0,      n0 + l15,      bias, outf, outb, resid, N, hf);
  epilogue_tile<MODE>(acc01, m0,      n0 + 16 + l15, bias, outf, outb, resid, N, hf);
  epilogue_tile<MODE>(acc10, m0 + 16, n0 + l15,      bias, outf, outb, resid, N, hf);
  epilogue_tile<MODE>(acc11, m0 + 16, n0 + 16 + l15, bias, outf, outb, resid, N, hf);
}

// ---- GEGLU FFN front half: h = a * gelu(g), p = hx@ff1 + b, a=p[:,:2048], g=p[:,2048:] ----
// Wave tile 32 rows x 32 a-cols (+ the paired g-cols): 8 WMMAs per K-step.
__global__ __launch_bounds__(256) void gemm_geglu_kernel(const bf16* __restrict__ A,
                                                         const bf16* __restrict__ Bt,   // [4096,512]
                                                         const float* __restrict__ bias, // [4096]
                                                         bf16* __restrict__ outh,        // [T,2048]
                                                         int K, int lda, int ldbt) {
  int lane = threadIdx.x & 31, wave = threadIdx.x >> 5;
  int hf = lane >> 4, l15 = lane & 15;
  int m0 = blockIdx.x * 128 + (wave >> 1) * 32;
  int n0 = blockIdx.y * 64 + (wave & 1) * 32;     // a-col base in [0,2048)
  v8f aa00 = {}; v8f aa01 = {}; v8f aa10 = {}; v8f aa11 = {};
  v8f gg00 = {}; v8f gg01 = {}; v8f gg10 = {}; v8f gg11 = {};
  for (int k0 = 0; k0 < K; k0 += 32) {
    v16bf a0 = load_a_frag(A, lda, m0 + l15,      k0, hf);
    v16bf a1 = load_a_frag(A, lda, m0 + 16 + l15, k0, hf);
    v16bf ba0 = load_bt_frag(Bt, ldbt, n0 + l15,             k0, hf);
    v16bf ba1 = load_bt_frag(Bt, ldbt, n0 + 16 + l15,        k0, hf);
    v16bf bg0 = load_bt_frag(Bt, ldbt, n0 + DFFH + l15,      k0, hf);
    v16bf bg1 = load_bt_frag(Bt, ldbt, n0 + DFFH + 16 + l15, k0, hf);
    aa00 = wmma_bf16(a0, ba0, aa00);
    aa01 = wmma_bf16(a0, ba1, aa01);
    aa10 = wmma_bf16(a1, ba0, aa10);
    aa11 = wmma_bf16(a1, ba1, aa11);
    gg00 = wmma_bf16(a0, bg0, gg00);
    gg01 = wmma_bf16(a0, bg1, gg01);
    gg10 = wmma_bf16(a1, bg0, gg10);
    gg11 = wmma_bf16(a1, bg1, gg11);
  }
#pragma unroll
  for (int mi = 0; mi < 2; ++mi) {
#pragma unroll
    for (int ni = 0; ni < 2; ++ni) {
      const v8f& av = mi ? (ni ? aa11 : aa10) : (ni ? aa01 : aa00);
      const v8f& gv = mi ? (ni ? gg11 : gg10) : (ni ? gg01 : gg00);
      int col = n0 + ni * 16 + l15;
#pragma unroll
      for (int r = 0; r < 8; ++r) {
        int m = m0 + mi * 16 + r + 8 * hf;
        float a = av[r] + bias[col];
        float g = gv[r] + bias[col + DFFH];
        float gl = 0.5f * g * (1.f + erff(g * 0.70710678118654752f));
        outh[(size_t)m * DFFH + col] = (bf16)(a * gl);
      }
    }
  }
}

// ---- Flash attention: one wave = 16 query rows of one (b,h). S = Q*K^T is one
// 16x16x32 bf16 WMMA (DH=32); online softmax via per-wave LDS; P*V via 2 WMMAs/chunk. ----
__global__ __launch_bounds__(128) void attention_kernel(const bf16* __restrict__ q,
                                                        const bf16* __restrict__ k,
                                                        const bf16* __restrict__ vt,
                                                        bf16* __restrict__ o) {
  __shared__ float s_s[4][16][32];
  __shared__ __align__(16) bf16 s_p[4][16][32];
  __shared__ float s_corr[4][16];
  __shared__ float s_l[4][16];
  int lane = threadIdx.x & 31, w = threadIdx.x >> 5;
  int hf = lane >> 4, l15 = lane & 15;
  int h = blockIdx.y, b = blockIdx.z;
  int i0 = (blockIdx.x * 4 + w) * 16;
  const bf16* qh = q  + ((size_t)(b * NHEAD + h)) * SEQ * DHEAD;
  const bf16* kh = k  + ((size_t)(b * NHEAD + h)) * SEQ * DHEAD;
  const bf16* vh = vt + ((size_t)(b * NHEAD + h)) * DHEAD * SEQ;
  v16bf qa = load_a_frag(qh, DHEAD, i0 + l15, 0, hf);
  v8f o0 = {}; v8f o1 = {};
  float mrow = -1e30f, lrow = 0.f;
  const float scale = 0.17677669529663687f;   // 1/sqrt(32)
  for (int j0 = 0; j0 < SEQ; j0 += 32) {
    v16bf kb0 = load_bt_frag(kh, DHEAD, j0 + l15, 0, hf);
    v16bf kb1 = load_bt_frag(kh, DHEAD, j0 + 16 + l15, 0, hf);
    v8f z = {};
    v8f s0 = wmma_bf16(qa, kb0, z);
    v8f s1 = wmma_bf16(qa, kb1, z);
#pragma unroll
    for (int r = 0; r < 8; ++r) {
      int rr = r + 8 * hf;
      s_s[w][rr][l15]      = s0[r] * scale;
      s_s[w][rr][l15 + 16] = s1[r] * scale;
    }
    __syncthreads();
    if (lane < 16) {
      float mx = mrow;
#pragma unroll
      for (int c = 0; c < 32; ++c) mx = fmaxf(mx, s_s[w][lane][c]);
      float corr = __expf(mrow - mx);
      float sum = 0.f;
#pragma unroll
      for (int c = 0; c < 32; ++c) {
        float e = __expf(s_s[w][lane][c] - mx);
        sum += e;
        s_p[w][lane][c] = (bf16)e;
      }
      lrow = lrow * corr + sum;
      mrow = mx;
      s_corr[w][lane] = corr;
    }
    __syncthreads();
#pragma unroll
    for (int r = 0; r < 8; ++r) {
      float cr = s_corr[w][r + 8 * hf];
      o0[r] *= cr; o1[r] *= cr;
    }
    v16bf pa  = load_a_frag(&s_p[w][0][0], 32, l15, 0, hf);
    v16bf vb0 = load_bt_frag(vh, SEQ, l15, j0, hf);
    v16bf vb1 = load_bt_frag(vh, SEQ, 16 + l15, j0, hf);
    o0 = wmma_bf16(pa, vb0, o0);
    o1 = wmma_bf16(pa, vb1, o1);
  }
  if (lane < 16) s_l[w][lane] = lrow;
  __syncthreads();
#pragma unroll
  for (int r = 0; r < 8; ++r) {
    int rr = r + 8 * hf;
    float inv = 1.0f / s_l[w][rr];
    size_t base = ((size_t)b * SEQ + i0 + rr) * CH + h * DHEAD;
    o[base + l15]      = (bf16)(o0[r] * inv);
    o[base + l15 + 16] = (bf16)(o1[r] * inv);
  }
}

extern "C" void kernel_launch(void* const* d_in, const int* in_sizes, int n_in,
                              void* d_out, int out_size, void* d_ws, size_t ws_size,
                              hipStream_t stream) {
  (void)in_sizes; (void)n_in; (void)out_size; (void)ws_size;
  const float* x     = (const float*)d_in[0];
  const float* gn_w  = (const float*)d_in[1];
  const float* gn_b  = (const float*)d_in[2];
  const float* pin_w = (const float*)d_in[3];
  const float* pin_b = (const float*)d_in[4];
  const float* pout_w= (const float*)d_in[5];
  const float* pout_b= (const float*)d_in[6];
  const float* ln1_w = (const float*)d_in[7];
  const float* ln1_b = (const float*)d_in[8];
  const float* wq    = (const float*)d_in[9];
  const float* wk    = (const float*)d_in[10];
  const float* wv    = (const float*)d_in[11];
  const float* wo    = (const float*)d_in[12];
  const float* bo    = (const float*)d_in[13];
  const float* ln3_w = (const float*)d_in[14];
  const float* ln3_b = (const float*)d_in[15];
  const float* ff1_w = (const float*)d_in[16];
  const float* ff1_b = (const float*)d_in[17];
  const float* ff2_w = (const float*)d_in[18];
  const float* ff2_b = (const float*)d_in[19];
  float* out = (float*)d_out;

  char* ws = (char*)d_ws;
  size_t off = 0;
  auto alloc = [&](size_t bytes) -> void* {
    void* p = ws + off;
    off = (off + bytes + 255) & ~(size_t)255;
    return p;
  };
  float* X     = (float*)alloc((size_t)TOKENS * CH * 4);           // residual stream f32
  bf16* tmp    = (bf16*)alloc((size_t)TOKENS * DFFH * 2);          // o (4MB) / geglu h (16MB)
  bf16* act    = (bf16*)alloc((size_t)TOKENS * CH * 2);            // gn / hx bf16
  bf16* qb     = (bf16*)alloc((size_t)TOKENS * CH * 2);
  bf16* kb     = (bf16*)alloc((size_t)TOKENS * CH * 2);
  bf16* vtb    = (bf16*)alloc((size_t)TOKENS * CH * 2);
  bf16* pinT   = (bf16*)alloc((size_t)CH * CH * 2);
  bf16* poutT  = (bf16*)alloc((size_t)CH * CH * 2);
  bf16* wqT    = (bf16*)alloc((size_t)NLAYER * CH * CH * 2);
  bf16* wkT    = (bf16*)alloc((size_t)NLAYER * CH * CH * 2);
  bf16* wvT    = (bf16*)alloc((size_t)NLAYER * CH * CH * 2);
  bf16* woT    = (bf16*)alloc((size_t)NLAYER * CH * CH * 2);
  bf16* ff1T   = (bf16*)alloc((size_t)NLAYER * 4096 * CH * 2);
  bf16* ff2T   = (bf16*)alloc((size_t)NLAYER * CH * DFFH * 2);

  // ---- weight conversion (transpose to [N,K] bf16) ----
  convert_wT_kernel<<<1024, 256, 0, stream>>>(pin_w,  pinT,  CH, CH, (long)CH * CH);
  convert_wT_kernel<<<1024, 256, 0, stream>>>(pout_w, poutT, CH, CH, (long)CH * CH);
  long qkvo_tot = (long)NLAYER * CH * CH;
  convert_wT_kernel<<<4096, 256, 0, stream>>>(wq, wqT, CH, CH, qkvo_tot);
  convert_wT_kernel<<<4096, 256, 0, stream>>>(wk, wkT, CH, CH, qkvo_tot);
  convert_wT_kernel<<<4096, 256, 0, stream>>>(wv, wvT, CH, CH, qkvo_tot);
  convert_wT_kernel<<<4096, 256, 0, stream>>>(wo, woT, CH, CH, qkvo_tot);
  convert_wT_kernel<<<32768, 256, 0, stream>>>(ff1_w, ff1T, CH, 4096, (long)NLAYER * CH * 4096);
  convert_wT_kernel<<<16384, 256, 0, stream>>>(ff2_w, ff2T, DFFH, CH, (long)NLAYER * DFFH * CH);

  // ---- GroupNorm + conv-in (proj_in) ----
  groupnorm_kernel<<<64, 256, 0, stream>>>(x, gn_w, gn_b, act);
  dim3 gC(TOKENS / 128, CH / 64);
  gemm_bf16_kernel<0><<<gC, 256, 0, stream>>>(act, pinT, pin_b, X, nullptr, nullptr, CH, CH, CH, CH);

  // ---- transformer layers ----
  for (int l = 0; l < NLAYER; ++l) {
    size_t wOff  = (size_t)l * CH * CH;
    layernorm_kernel<<<TOKENS, 128, 0, stream>>>(X, ln1_w + l * CH, ln1_b + l * CH, act);
    gemm_bf16_kernel<1><<<gC, 256, 0, stream>>>(act, wqT + wOff, nullptr, nullptr, qb,  nullptr, CH, CH, CH, CH);
    gemm_bf16_kernel<1><<<gC, 256, 0, stream>>>(act, wkT + wOff, nullptr, nullptr, kb,  nullptr, CH, CH, CH, CH);
    gemm_bf16_kernel<2><<<gC, 256, 0, stream>>>(act, wvT + wOff, nullptr, nullptr, vtb, nullptr, CH, CH, CH, CH);
    attention_kernel<<<dim3(SEQ / 64, NHEAD, 4), 128, 0, stream>>>(qb, kb, vtb, tmp);
    gemm_bf16_kernel<3><<<gC, 256, 0, stream>>>(tmp, woT + wOff, bo + l * CH, X, nullptr, nullptr, CH, CH, CH, CH);

    layernorm_kernel<<<TOKENS, 128, 0, stream>>>(X, ln3_w + l * CH, ln3_b + l * CH, act);
    gemm_geglu_kernel<<<dim3(TOKENS / 128, DFFH / 64), 256, 0, stream>>>(
        act, ff1T + (size_t)l * 4096 * CH, ff1_b + (size_t)l * 4096, tmp, CH, CH, CH);
    gemm_bf16_kernel<3><<<gC, 256, 0, stream>>>(tmp, ff2T + (size_t)l * CH * DFFH, ff2_b + l * CH,
                                                X, nullptr, nullptr, CH, DFFH, DFFH, DFFH);
  }

  // ---- conv-out (proj_out) + input residual ----
  f32_to_bf16_kernel<<<(TOKENS * CH) / 256, 256, 0, stream>>>(X, act, TOKENS * CH);
  gemm_bf16_kernel<4><<<gC, 256, 0, stream>>>(act, poutT, pout_b, out, nullptr, x, CH, CH, CH, CH);
}
